// BitAttention_28252294873582
// MI455X (gfx1250) — 16-bit and 8-bit matrix operand order repaired and checked by reading, compiled, not run on hardware
//
#include <hip/hip_runtime.h>
#include <hip/hip_bf16.h>
#include <math.h>


typedef __attribute__((ext_vector_type(16))) _Float16 v16h;
typedef __attribute__((ext_vector_type(8)))  float    v8f;
typedef __attribute__((ext_vector_type(8)))  int      v8i;
typedef __attribute__((ext_vector_type(2)))  _Float16 h2;

#define BB   2
#define SS   2048
#define DD   2048
#define HH   16
#define KHH  4
#define HDD  128
#define KVD  512
#define MM   (BB*SS)

__device__ __forceinline__ void async_load_b128(unsigned lds_off, const void* gaddr) {
    asm volatile("global_load_async_to_lds_b128 %0, %1, off"
                 :: "v"(lds_off), "v"((unsigned long long)(size_t)gaddr)
                 : "memory");
}
__device__ __forceinline__ void wait_async0() {
    asm volatile("s_wait_asynccnt 0" ::: "memory");
}


__global__ void zero_accum(float* a) {
    if (threadIdx.x < 8) a[threadIdx.x] = 0.f;
}

__global__ __launch_bounds__(256)
void wabs_reduce(const float* __restrict__ w, long long n, float* __restrict__ accum) {
    __shared__ float red[8];
    float s = 0.f;
    for (long long i = (long long)blockIdx.x * 256 + threadIdx.x; i < n;
         i += (long long)gridDim.x * 256)
        s += fabsf(w[i]);
    for (int d = 16; d; d >>= 1) s += __shfl_xor(s, d, 32);
    if ((threadIdx.x & 31) == 0) red[threadIdx.x >> 5] = s;
    __syncthreads();
    if (threadIdx.x == 0) {
        float t = 0.f;
        #pragma unroll
        for (int i = 0; i < 8; ++i) t += red[i];
        atomicAdd(accum, t);
    }
}

__global__ __launch_bounds__(256)
void wquant(const float* __restrict__ w, signed char* __restrict__ wq,
            long long n, const float* __restrict__ sumabs, float* __restrict__ wscale) {
    float s = (*sumabs) / (float)n;
    long long i = (long long)blockIdx.x * 256 + threadIdx.x;
    if (i < n) wq[i] = (signed char)rintf(tanhf(w[i] / (s + 1e-8f)));
    if (blockIdx.x == 0 && threadIdx.x == 0) *wscale = atanhf(s);
}


__global__ __launch_bounds__(256)
void rms_quant(const float* __restrict__ x, const float* __restrict__ g,
               signed char* __restrict__ xq, float* __restrict__ aScale, int W) {
    __shared__ float red[8];
    const int row = blockIdx.x, tid = threadIdx.x;
    const float* xr = x + (size_t)row * W;
    float vals[8];
    float ss = 0.f; int cnt = 0;
    for (int i = tid; i < W; i += 256) { float v = xr[i]; vals[cnt++] = v; ss += v * v; }
    for (int d = 16; d; d >>= 1) ss += __shfl_xor(ss, d, 32);
    if ((tid & 31) == 0) red[tid >> 5] = ss;
    __syncthreads();
    if (tid == 0) { float t = 0.f; for (int i = 0; i < 8; ++i) t += red[i]; red[0] = t; }
    __syncthreads();
    const float rinv = rsqrtf(red[0] / (float)W + 1e-8f);
    __syncthreads();
    float mx = 0.f; cnt = 0;
    for (int i = tid; i < W; i += 256) {
        float nv = vals[cnt++] * rinv * g[i];
        mx = fmaxf(mx, fabsf(nv));
    }
    for (int d = 16; d; d >>= 1) mx = fmaxf(mx, __shfl_xor(mx, d, 32));
    if ((tid & 31) == 0) red[tid >> 5] = mx;
    __syncthreads();
    if (tid == 0) { float t = 0.f; for (int i = 0; i < 8; ++i) t = fmaxf(t, red[i]); red[0] = t; }
    __syncthreads();
    const float sc = 127.f / fmaxf(red[0], 1e-4f);
    cnt = 0;
    for (int i = tid; i < W; i += 256) {
        float nv = vals[cnt++] * rinv * g[i];
        float q = fminf(127.f, fmaxf(-128.f, rintf(nv * sc)));
        xq[(size_t)row * W + i] = (signed char)q;
    }
    if (tid == 0) aScale[row] = 1.f / sc;
}


#define Bb_M 64
#define Bb_N 128
#define Bb_K 128

__global__ __launch_bounds__(256)
void gemm_iu8(const signed char* __restrict__ A, const signed char* __restrict__ W,
              float* __restrict__ out, int M, int N, int K,
              const float* __restrict__ aScale, const float* __restrict__ wScalePtr) {
    __shared__ __align__(16) signed char As[2][Bb_M * Bb_K];
    __shared__ __align__(16) signed char Bs[2][Bb_N * Bb_K];
    const int tid  = threadIdx.x;
    const int lane = tid & 31, wid = tid >> 5;
    const int wm = wid & 3, wn = wid >> 2;
    const int half = lane >> 4, ml = lane & 15;
    const int m0 = blockIdx.y * Bb_M, n0 = blockIdx.x * Bb_N;

    const int aIdx = tid * 32;
    const int aRow = aIdx >> 7, aCol = aIdx & 127;
    const int bIdx = tid * 64;
    const int bRow = bIdx >> 7, bCol = bIdx & 127;
    const signed char* aSrc = &A[(size_t)(m0 + aRow) * K + aCol];
    const signed char* bSrc = &W[(size_t)(n0 + bRow) * K + bCol];

    v8i acc[4];
    #pragma unroll
    for (int f = 0; f < 4; ++f) acc[f] = (v8i){0, 0, 0, 0, 0, 0, 0, 0};

    async_load_b128((unsigned)(size_t)&As[0][aIdx],      aSrc);
    async_load_b128((unsigned)(size_t)&As[0][aIdx + 16], aSrc + 16);
    #pragma unroll
    for (int i = 0; i < 4; ++i)
        async_load_b128((unsigned)(size_t)&Bs[0][bIdx + 16 * i], bSrc + 16 * i);

    int cur = 0;
    for (int k0 = 0; k0 < K; k0 += Bb_K, cur ^= 1) {
        wait_async0();
        __syncthreads();
        if (k0 + Bb_K < K) {
            const int nk = k0 + Bb_K;
            async_load_b128((unsigned)(size_t)&As[cur ^ 1][aIdx],      aSrc + nk);
            async_load_b128((unsigned)(size_t)&As[cur ^ 1][aIdx + 16], aSrc + nk + 16);
            #pragma unroll
            for (int i = 0; i < 4; ++i)
                async_load_b128((unsigned)(size_t)&Bs[cur ^ 1][bIdx + 16 * i],
                                bSrc + nk + 16 * i);
        }
        const signed char* ap = &As[cur][(16 * wm + ml) * Bb_K];
        v8i af[2];
        #pragma unroll
        for (int kh = 0; kh < 2; ++kh)
            #pragma unroll
            for (int v = 0; v < 8; ++v) {
                int ka = kh * 64 + ((v >> 2) << 5) + ((v & 3) << 2) + (half << 4);
                af[kh][v] = *(const int*)(ap + ka);
            }
        v8i bf[8];
        #pragma unroll
        for (int f = 0; f < 4; ++f) {
            const signed char* bp = &Bs[cur][(wn * 64 + f * 16 + ml) * Bb_K];
            #pragma unroll
            for (int kh = 0; kh < 2; ++kh)
                #pragma unroll
                for (int v = 0; v < 8; ++v) {
                    int kb = kh * 64 + ((v >> 2) << 5) + ((v & 3) << 2) + (half << 4);
                    bf[kh * 4 + f][v] = *(const int*)(bp + kb);
                }
        }
        #pragma unroll
        for (int f = 0; f < 4; ++f)
            acc[f] = __builtin_amdgcn_wmma_i32_16x16x64_iu8(true, af[0], true, bf[f],
                                                            acc[f], false, false);
        #pragma unroll
        for (int f = 0; f < 4; ++f)
            acc[f] = __builtin_amdgcn_wmma_i32_16x16x64_iu8(true, af[1], true, bf[4 + f],
                                                            acc[f], false, false);
    }
    const float ws = *wScalePtr;
    #pragma unroll
    for (int r = 0; r < 8; ++r) {
        const int row = m0 + 16 * wm + r + 8 * half;
        const float as = aScale[row] * ws;
        #pragma unroll
        for (int f = 0; f < 4; ++f) {
            const int col = n0 + wn * 64 + f * 16 + ml;
            out[(size_t)row * N + col] = (float)acc[f][r] * as;
        }
    }
}


__global__ __launch_bounds__(256)
void rope_pack(const float* __restrict__ qr, const float* __restrict__ kr,
               const float* __restrict__ vr, const float* __restrict__ cosb,
               const float* __restrict__ sinb, _Float16* __restrict__ qs,
               _Float16* __restrict__ kf, _Float16* __restrict__ vf) {
    int n = blockIdx.x * 256 + threadIdx.x;
    if (n >= BB * KHH * SS * 64) return;
    const int j = n & 63;
    const int s = (n >> 6) & (SS - 1);
    const int h = (n >> 17) & 3;
    const int b = n >> 19;
    const float c  = cosb[s * 64 + j];
    const float sn = sinb[s * 64 + j];
    const size_t row = (size_t)b * SS + s;
    float x0 = 0.f, x1 = 0.f;
    #pragma unroll
    for (int gi = 0; gi < 4; ++gi) {
        size_t base = row * DD + (size_t)(gi * 4 + h) * HDD + 2 * j;
        x0 += qr[base]; x1 += qr[base + 1];
    }
    const size_t o = ((size_t)(b * KHH + h) * SS + s) * HDD + 2 * j;
    qs[o]     = (_Float16)(x0 * c - x1 * sn);
    qs[o + 1] = (_Float16)(x0 * sn + x1 * c);
    const size_t kb = row * KVD + (size_t)h * HDD + 2 * j;
    const float k0 = kr[kb], k1 = kr[kb + 1];
    kf[o]     = (_Float16)(k0 * c - k1 * sn);
    kf[o + 1] = (_Float16)(k0 * sn + k1 * c);
    vf[o]     = (_Float16)vr[kb];
    vf[o + 1] = (_Float16)vr[kb + 1];
}


__global__ __launch_bounds__(128)
void attn(const _Float16* __restrict__ qs, const _Float16* __restrict__ kf,
          const _Float16* __restrict__ vf, const int* __restrict__ causalp,
          float* __restrict__ ctx) {
    __shared__ _Float16 Plds[4][16][32];
    const int tid = threadIdx.x, lane = tid & 31, wid = tid >> 5;
    const int w  = blockIdx.x * 4 + wid;
    const int qb = w & 127;
    const int h  = (w >> 7) & 3;
    const int b  = w >> 9;
    const int half = lane >> 4, nl = lane & 15;
    const int causal = causalp[0];
    const size_t bh = (size_t)(b * KHH + h) * SS;

    v16h qf[4];
    const _Float16* qrow = qs + (bh + qb * 16 + nl) * HDD;
    #pragma unroll
    for (int c = 0; c < 4; ++c) {
        #pragma unroll
        for (int v = 0; v < 8; ++v) {
            int kp = c * 32 + 2 * v + 16 * half;
            h2 t = *(const h2*)(qrow + kp);
            qf[c][2 * v] = t[0]; qf[c][2 * v + 1] = t[1];
        }
    }
    v8f o[8];
    #pragma unroll
    for (int f = 0; f < 8; ++f) o[f] = (v8f){0, 0, 0, 0, 0, 0, 0, 0};
    float rmax[8], lsum[8];
    #pragma unroll
    for (int r = 0; r < 8; ++r) { rmax[r] = -1e38f; lsum[r] = 0.f; }

    const int limit = causal ? (qb * 16 + 16) : SS;
    for (int kc = 0; kc < limit; kc += 32) {
        v8f s0 = {0, 0, 0, 0, 0, 0, 0, 0};
        v8f s1 = {0, 0, 0, 0, 0, 0, 0, 0};
        #pragma unroll
        for (int c = 0; c < 4; ++c) {
            const _Float16* kp0 = kf + (bh + kc + nl) * HDD + c * 32 + 16 * half;
            const _Float16* kp1 = kp0 + (size_t)16 * HDD;
            v16h b0, b1;
            #pragma unroll
            for (int v = 0; v < 8; ++v) {
                h2 t0 = *(const h2*)(kp0 + 2 * v);
                h2 t1 = *(const h2*)(kp1 + 2 * v);
                b0[2 * v] = t0[0]; b0[2 * v + 1] = t0[1];
                b1[2 * v] = t1[0]; b1[2 * v + 1] = t1[1];
            }
            s0 = __builtin_amdgcn_wmma_f32_16x16x32_f16(false, qf[c], false, b0,
                                                        (short)0, s0, false, false);
            s1 = __builtin_amdgcn_wmma_f32_16x16x32_f16(false, qf[c], false, b1,
                                                        (short)0, s1, false, false);
        }
        const float invhd = 1.0f / 128.0f;
        #pragma unroll
        for (int r = 0; r < 8; ++r) {
            const int qidx = qb * 16 + r + 8 * half;
            float a0 = s0[r] * invhd, a1 = s1[r] * invhd;
            if (causal) {
                if (kc + nl      > qidx) a0 = -1e30f;
                if (kc + 16 + nl > qidx) a1 = -1e30f;
            }
            float m = fmaxf(a0, a1);
            #pragma unroll
            for (int d = 1; d < 16; d <<= 1) m = fmaxf(m, __shfl_xor(m, d, 32));
            const float nm  = fmaxf(rmax[r], m);
            const float fac = expf(rmax[r] - nm);
            rmax[r] = nm;
            const float p0 = expf(a0 - nm), p1 = expf(a1 - nm);
            s0[r] = p0; s1[r] = p1;
            float rs = p0 + p1;
            #pragma unroll
            for (int d = 1; d < 16; d <<= 1) rs += __shfl_xor(rs, d, 32);
            lsum[r] = lsum[r] * fac + rs;
            #pragma unroll
            for (int f = 0; f < 8; ++f) o[f][r] *= fac;
        }
        #pragma unroll
        for (int r = 0; r < 8; ++r) {
            int m = r + 8 * half;
            Plds[wid][m][nl]      = (_Float16)s0[r];
            Plds[wid][m][nl + 16] = (_Float16)s1[r];
        }
        v16h pf;
        #pragma unroll
        for (int v = 0; v < 8; ++v) {
            int kp = 2 * v + 16 * half;
            pf[2 * v]     = Plds[wid][nl][kp];
            pf[2 * v + 1] = Plds[wid][nl][kp + 1];
        }
        #pragma unroll
        for (int f = 0; f < 8; ++f) {
            v16h vfr;
            #pragma unroll
            for (int v = 0; v < 8; ++v) {
                int k0 = 2 * v + 16 * half;
                vfr[2 * v]     = vf[(bh + kc + k0)     * HDD + f * 16 + nl];
                vfr[2 * v + 1] = vf[(bh + kc + k0 + 1) * HDD + f * 16 + nl];
            }
            o[f] = __builtin_amdgcn_wmma_f32_16x16x32_f16(false, pf, false, vfr,
                                                          (short)0, o[f], false, false);
        }
    }
    #pragma unroll
    for (int r = 0; r < 8; ++r) {
        const float inv = 1.0f / lsum[r];
        const int m = qb * 16 + r + 8 * half;
        const size_t rowbase = ((size_t)b * SS + m) * KVD + (size_t)h * HDD;
        #pragma unroll
        for (int f = 0; f < 8; ++f)
            ctx[rowbase + f * 16 + nl] = o[f][r] * inv;
    }
}


extern "C" void kernel_launch(void* const* d_in, const int* in_sizes, int n_in,
                              void* d_out, int out_size, void* d_ws, size_t ws_size,
                              hipStream_t stream) {
    (void)in_sizes; (void)n_in; (void)out_size; (void)ws_size;
    const float* x    = (const float*)d_in[0];
    const float* fcos = (const float*)d_in[1];
    const float* fsin = (const float*)d_in[2];
    const float* w_q  = (const float*)d_in[3];
    const float* w_k  = (const float*)d_in[4];
    const float* w_v  = (const float*)d_in[5];
    const float* w_o  = (const float*)d_in[6];
    const float* g_q  = (const float*)d_in[7];
    const float* g_k  = (const float*)d_in[8];
    const float* g_v  = (const float*)d_in[9];
    const float* g_o  = (const float*)d_in[10];
    const int*   causal = (const int*)d_in[11];
    float* out = (float*)d_out;

    char* p = (char*)d_ws;
    auto alloc = [&](size_t bytes) -> char* {
        char* r = p; p += (bytes + 255) & ~(size_t)255; return r;
    };
    float*       acc    = (float*)alloc(8 * sizeof(float));
    signed char* xq_q   = (signed char*)alloc((size_t)MM * DD);
    signed char* xq_k   = (signed char*)alloc((size_t)MM * DD);
    signed char* xq_v   = (signed char*)alloc((size_t)MM * DD);
    float*       asc_q  = (float*)alloc(MM * sizeof(float));
    float*       asc_k  = (float*)alloc(MM * sizeof(float));
    float*       asc_v  = (float*)alloc(MM * sizeof(float));
    signed char* wq_q   = (signed char*)alloc((size_t)DD * DD);
    signed char* wq_k   = (signed char*)alloc((size_t)KVD * DD);
    signed char* wq_v   = (signed char*)alloc((size_t)KVD * DD);
    signed char* wq_o   = (signed char*)alloc((size_t)DD * KVD);
    float*       q_raw  = (float*)alloc((size_t)MM * DD * 4);
    float*       k_raw  = (float*)alloc((size_t)MM * KVD * 4);
    float*       v_raw  = (float*)alloc((size_t)MM * KVD * 4);
    _Float16*    qs     = (_Float16*)alloc((size_t)BB * KHH * SS * HDD * 2);
    _Float16*    kfp    = (_Float16*)alloc((size_t)BB * KHH * SS * HDD * 2);
    _Float16*    vfp    = (_Float16*)alloc((size_t)BB * KHH * SS * HDD * 2);
    float*       ctx    = (float*)alloc((size_t)MM * KVD * 4);
    signed char* ctxq   = (signed char*)alloc((size_t)MM * KVD);
    float*       csc    = (float*)alloc(MM * sizeof(float));

    zero_accum<<<1, 32, 0, stream>>>(acc);
    const long long nQ = (long long)DD * DD, nK = (long long)KVD * DD,
                    nO = (long long)DD * KVD;
    wabs_reduce<<<512, 256, 0, stream>>>(w_q, nQ, &acc[0]);
    wabs_reduce<<<512, 256, 0, stream>>>(w_k, nK, &acc[1]);
    wabs_reduce<<<512, 256, 0, stream>>>(w_v, nK, &acc[2]);
    wabs_reduce<<<512, 256, 0, stream>>>(w_o, nO, &acc[3]);
    wquant<<<(unsigned)((nQ + 255) / 256), 256, 0, stream>>>(w_q, wq_q, nQ, &acc[0], &acc[4]);
    wquant<<<(unsigned)((nK + 255) / 256), 256, 0, stream>>>(w_k, wq_k, nK, &acc[1], &acc[5]);
    wquant<<<(unsigned)((nK + 255) / 256), 256, 0, stream>>>(w_v, wq_v, nK, &acc[2], &acc[6]);
    wquant<<<(unsigned)((nO + 255) / 256), 256, 0, stream>>>(w_o, wq_o, nO, &acc[3], &acc[7]);

    rms_quant<<<MM, 256, 0, stream>>>(x, g_q, xq_q, asc_q, DD);
    rms_quant<<<MM, 256, 0, stream>>>(x, g_k, xq_k, asc_k, DD);
    rms_quant<<<MM, 256, 0, stream>>>(x, g_v, xq_v, asc_v, DD);

    gemm_iu8<<<dim3(DD / Bb_N,  MM / Bb_M), 256, 0, stream>>>(xq_q, wq_q, q_raw, MM, DD,  DD, asc_q, &acc[4]);
    gemm_iu8<<<dim3(KVD / Bb_N, MM / Bb_M), 256, 0, stream>>>(xq_k, wq_k, k_raw, MM, KVD, DD, asc_k, &acc[5]);
    gemm_iu8<<<dim3(KVD / Bb_N, MM / Bb_M), 256, 0, stream>>>(xq_v, wq_v, v_raw, MM, KVD, DD, asc_v, &acc[6]);

    rope_pack<<<(BB * KHH * SS * 64) / 256, 256, 0, stream>>>(
        q_raw, k_raw, v_raw, fcos, fsin, qs, kfp, vfp);

    attn<<<(BB * KHH * (SS / 16)) / 4, 128, 0, stream>>>(qs, kfp, vfp, causal, ctx);

    rms_quant<<<MM, 256, 0, stream>>>(ctx, g_o, ctxq, csc, KVD);
    gemm_iu8<<<dim3(DD / Bb_N, MM / Bb_M), 256, 0, stream>>>(ctxq, wq_o, out, MM, DD, KVD, csc, &acc[7]);
}
